// GraphClassifier_54357106098294
// MI455X (gfx1250) — matrix operand order repaired and checked by reading, not compiled, not run on hardware
//
#include <hip/hip_runtime.h>
#include <hip/hip_bf16.h>

#define NNODES 50000
#define NEDGES 800000
#define FDIM   128

typedef __attribute__((ext_vector_type(16))) __bf16 v16bf;
typedef __attribute__((ext_vector_type(8)))  float  v8f;

// ---------- helpers ----------
__device__ __forceinline__ unsigned short f2bf(float x) {
  union { float f; unsigned u; } v; v.f = x;
  unsigned r = v.u + 0x7FFFu + ((v.u >> 16) & 1u);   // round-to-nearest-even
  return (unsigned short)(r >> 16);
}

__global__ void zero_f32(float* __restrict__ p, int n) {
  for (int i = blockIdx.x * blockDim.x + threadIdx.x; i < n; i += gridDim.x * blockDim.x)
    p[i] = 0.0f;
}

// ---------- degree / norm ----------
__global__ void degree_kernel(const int* __restrict__ ei, float* __restrict__ deg) {
  int e = blockIdx.x * blockDim.x + threadIdx.x;
  if (e < NEDGES) atomicAdd(&deg[ei[NEDGES + e]], 1.0f);   // dst row of edge_index
}

__global__ void dinv_kernel(float* __restrict__ deg) {
  int i = blockIdx.x * blockDim.x + threadIdx.x;
  if (i < NNODES) deg[i] = rsqrtf(deg[i] + 1.0f);          // +1 for self loop
}

// ---------- precision conversion ----------
__global__ void f32_to_bf16(const float* __restrict__ s, unsigned short* __restrict__ d, int n) {
  for (int i = blockIdx.x * blockDim.x + threadIdx.x; i < n; i += gridDim.x * blockDim.x)
    d[i] = f2bf(s[i]);
}

// W [in=128, out=128] f32  ->  WT [out=128, in=128] bf16 (B-fragment friendly)
__global__ void transpose_w_bf16(const float* __restrict__ w, unsigned short* __restrict__ wT) {
  int idx = blockIdx.x * blockDim.x + threadIdx.x;         // 16384 threads
  int i = idx >> 7, o = idx & 127;
  wT[o * FDIM + i] = f2bf(w[i * FDIM + o]);
}

// ---------- WMMA GEMM: H[N x 128] = A_bf16[N x 128] * WT_bf16^T ----------
// 8 waves/block; wave w owns the 16x16 C tile at (blockIdx.x*16, w*16). K=128 -> 4 wmma.
__global__ __launch_bounds__(256) void gemm_bf16_wmma(const unsigned short* __restrict__ A,
                                                      const unsigned short* __restrict__ BT,
                                                      float* __restrict__ H) {
  const int m0   = blockIdx.x * 16;
  const int wave = threadIdx.x >> 5;
  const int lane = threadIdx.x & 31;
  const int n0   = wave * 16;
  const int l15  = lane & 15;
  const int hi   = lane >> 4;

  const unsigned short* arow = A  + (size_t)(m0 + l15) * FDIM;   // A row M = m0 + (lane&15)
  const unsigned short* brow = BT + (size_t)(n0 + l15) * FDIM;   // B col N = n0 + (lane&15)

  v8f c = {};
#pragma unroll
  for (int kb = 0; kb < 4; ++kb) {
    const int k0 = kb * 32;
    union { v16bf v; uint4 q[2]; } a, b;
    // A 16x32 bf16 layout: lanes 0-15 hold K {0..7, 16..23}; lanes 16-31 hold K {8..15, 24..31}
    a.q[0] = *(const uint4*)(arow + k0 + hi * 8);
    a.q[1] = *(const uint4*)(arow + k0 + 16 + hi * 8);
    // B 32x16 bf16 layout: lanes 0-15 hold K {0..7, 16..23}; lanes 16-31 hold K {8..15, 24..31}
    b.q[0] = *(const uint4*)(brow + k0 + hi * 8);
    b.q[1] = *(const uint4*)(brow + k0 + 16 + hi * 8);
    c = __builtin_amdgcn_wmma_f32_16x16x32_bf16(false, a.v, false, b.v, (short)0, c, false, false);
  }

  // C layout: lanes 0-15 -> rows M=r, lanes 16-31 -> rows M=r+8; column N = lane&15
  float* hrow = H + n0 + l15;
#pragma unroll
  for (int r = 0; r < 8; ++r)
    hrow[(size_t)(m0 + r + hi * 8) * FDIM] = c[r];
}

// ---------- edge message passing: agg[dst] += h[src] * dinv[src]*dinv[dst] ----------
// one wave per edge: 32 lanes x float4 = 128 features, coalesced 512B row per edge
__global__ __launch_bounds__(256) void edge_scatter(const int* __restrict__ ei,
                                                    const float* __restrict__ dinv,
                                                    const float* __restrict__ h,
                                                    float* __restrict__ agg) {
  const int e = blockIdx.x * 8 + (threadIdx.x >> 5);
  if (e >= NEDGES) return;
  const int lane = threadIdx.x & 31;
  const int src = ei[e];
  const int dst = ei[NEDGES + e];
  const float norm = dinv[src] * dinv[dst];
  const float4 m = *(const float4*)(h + (size_t)src * FDIM + lane * 4);
  float* o = agg + (size_t)dst * FDIM + lane * 4;
  atomicAdd(o + 0, m.x * norm);
  atomicAdd(o + 1, m.y * norm);
  atomicAdd(o + 2, m.z * norm);
  atomicAdd(o + 3, m.w * norm);
}

// ---------- self-loop + bias + ReLU; keep f32 in agg (for pooling) and bf16 (next GEMM) ----------
__global__ void finalize_conv(float* __restrict__ agg, const float* __restrict__ h,
                              const float* __restrict__ dinv, const float* __restrict__ bias,
                              unsigned short* __restrict__ outbf) {
  const int total = NNODES * FDIM;
  for (int idx = blockIdx.x * blockDim.x + threadIdx.x; idx < total; idx += gridDim.x * blockDim.x) {
    const int i = idx >> 7, f = idx & 127;
    const float di = dinv[i];
    float v = agg[idx] + h[idx] * di * di + bias[f];
    v = fmaxf(v, 0.0f);
    agg[idx]   = v;
    outbf[idx] = f2bf(v);
  }
}

// ---------- global mean pool (sum; divide by N in head) ----------
__global__ __launch_bounds__(256) void pool_kernel(const float* __restrict__ v, float* __restrict__ pooled) {
  __shared__ float s[256];
  float acc = 0.0f;
  const int total = NNODES * FDIM;
  for (int idx = blockIdx.x * 256 + threadIdx.x; idx < total; idx += gridDim.x * 256)
    acc += v[idx];                       // stride is multiple of 128 -> fixed feature per thread
  s[threadIdx.x] = acc;
  __syncthreads();
  if (threadIdx.x < 128)
    atomicAdd(&pooled[threadIdx.x], s[threadIdx.x] + s[threadIdx.x + 128]);
}

// ---------- MLP head: [1x128] through 4 linears + sigmoid ----------
__global__ __launch_bounds__(128) void head_kernel(const float* __restrict__ pooled,
                                                   const float* __restrict__ w11, const float* __restrict__ b11,
                                                   const float* __restrict__ w12, const float* __restrict__ b12,
                                                   const float* __restrict__ w21, const float* __restrict__ b21,
                                                   const float* __restrict__ w22, const float* __restrict__ b22,
                                                   float* __restrict__ out) {
  __shared__ float g[128], t[128], red[128];
  const int j = threadIdx.x;
  g[j] = pooled[j] * (1.0f / (float)NNODES);
  __syncthreads();

  float s = b11[j];
  for (int i = 0; i < 128; ++i) s += g[i] * w11[i * 128 + j];
  t[j] = fmaxf(s, 0.0f);
  __syncthreads();

  s = b12[j];
  for (int i = 0; i < 128; ++i) s += t[i] * w12[i * 128 + j];
  g[j] = s;                                   // fc1 output: no ReLU (per reference)
  __syncthreads();

  s = b21[j];
  for (int i = 0; i < 128; ++i) s += g[i] * w21[i * 128 + j];
  t[j] = fmaxf(s, 0.0f);
  __syncthreads();

  red[j] = t[j] * w22[j];                     // fc2_w2 is [128,1]
  __syncthreads();
  for (int off = 64; off > 0; off >>= 1) {
    if (j < off) red[j] += red[j + off];
    __syncthreads();
  }
  if (j == 0) {
    const float y = red[0] + b22[0];
    out[0] = 1.0f / (1.0f + expf(-y));
  }
}

// ---------- launcher ----------
extern "C" void kernel_launch(void* const* d_in, const int* in_sizes, int n_in,
                              void* d_out, int out_size, void* d_ws, size_t ws_size,
                              hipStream_t stream) {
  const float* x    = (const float*)d_in[0];
  const int*   ei   = (const int*)  d_in[1];
  const float* c1w  = (const float*)d_in[2];
  const float* c1b  = (const float*)d_in[3];
  const float* c2w  = (const float*)d_in[4];
  const float* c2b  = (const float*)d_in[5];
  const float* f11w = (const float*)d_in[6];
  const float* f11b = (const float*)d_in[7];
  const float* f12w = (const float*)d_in[8];
  const float* f12b = (const float*)d_in[9];
  const float* f21w = (const float*)d_in[10];
  const float* f21b = (const float*)d_in[11];
  const float* f22w = (const float*)d_in[12];
  const float* f22b = (const float*)d_in[13];

  char* ws = (char*)d_ws;
  size_t off = 0;
  auto alloc = [&](size_t bytes) -> void* {
    void* p = ws + off;
    off = (off + bytes + 255) & ~(size_t)255;
    return p;
  };
  float*          dinv   = (float*)         alloc((size_t)NNODES * 4);
  float*          pooled = (float*)         alloc(128 * 4);
  unsigned short* xbf    = (unsigned short*)alloc((size_t)NNODES * FDIM * 2);
  unsigned short* w1T    = (unsigned short*)alloc((size_t)FDIM * FDIM * 2);
  unsigned short* w2T    = (unsigned short*)alloc((size_t)FDIM * FDIM * 2);
  float*          h      = (float*)         alloc((size_t)NNODES * FDIM * 4);
  float*          agg    = (float*)         alloc((size_t)NNODES * FDIM * 4);

  // degrees + symmetric norm
  zero_f32<<<256, 256, 0, stream>>>(dinv, NNODES);
  zero_f32<<<1, 128, 0, stream>>>(pooled, 128);
  degree_kernel<<<(NEDGES + 255) / 256, 256, 0, stream>>>(ei, dinv);
  dinv_kernel<<<(NNODES + 255) / 256, 256, 0, stream>>>(dinv);

  // precision prep
  f32_to_bf16<<<2048, 256, 0, stream>>>(x, xbf, NNODES * FDIM);
  transpose_w_bf16<<<64, 256, 0, stream>>>(c1w, w1T);
  transpose_w_bf16<<<64, 256, 0, stream>>>(c2w, w2T);

  // conv1
  gemm_bf16_wmma<<<NNODES / 16, 256, 0, stream>>>(xbf, w1T, h);
  zero_f32<<<2048, 256, 0, stream>>>(agg, NNODES * FDIM);
  edge_scatter<<<NEDGES / 8, 256, 0, stream>>>(ei, dinv, h, agg);
  finalize_conv<<<2048, 256, 0, stream>>>(agg, h, dinv, c1b, xbf);

  // conv2
  gemm_bf16_wmma<<<NNODES / 16, 256, 0, stream>>>(xbf, w2T, h);
  zero_f32<<<2048, 256, 0, stream>>>(agg, NNODES * FDIM);
  edge_scatter<<<NEDGES / 8, 256, 0, stream>>>(ei, dinv, h, agg);
  finalize_conv<<<2048, 256, 0, stream>>>(agg, h, dinv, c2b, xbf);

  // pool + head
  pool_kernel<<<1024, 256, 0, stream>>>(agg, pooled);
  head_kernel<<<1, 128, 0, stream>>>(pooled, f11w, f11b, f12w, f12b,
                                     f21w, f21b, f22w, f22b, (float*)d_out);
}
